// MHAMixer_73194832659004
// MI455X (gfx1250) — matrix operand order repaired and checked by reading, not compiled, not run on hardware
//
#include <hip/hip_runtime.h>

// ---------------------------------------------------------------------------
// MHAMixer for MI455X (gfx1250, wave32, WMMA).
//
// Pipeline (all f32 accumulation, f16 WMMA operands):
//   1) prep_weights : f32 weights -> f16, transposed to [N][K] for B-frags
//   2) proj_linear  : q/k/v @ lw  (+bias) -> qh/kh/vh heads 0..1 (f16)
//   3) proj_conv    : conv1d(k=3) as 3 shifted GEMMs -> heads 2..3 (f16)
//   4) logits       : att = (Q K^T) * 1/sqrt(128)  -> d_out att region (f32)
//                     K chunk staged in LDS via GLOBAL_LOAD_ASYNC_TO_LDS_B128
//                     (bank-conflict pad applied during the DMA itself)
//   5) softmax      : row softmax in place on att
//   6) attv         : hout = scale * att @ V  (att f32->f16 on the fly,
//                     V transposed through LDS)  -> hout (f16)
//   7) final_proj   : out = hout @ proj_w + proj_b -> d_out (f32)
//
// Workspace requirement: ~72 MB (71,303,168 bytes).
// ---------------------------------------------------------------------------

typedef _Float16 half_t;
typedef __attribute__((ext_vector_type(16))) _Float16 v16h;
typedef __attribute__((ext_vector_type(8)))  _Float16 v8h;
typedef __attribute__((ext_vector_type(8)))  float    v8f;
typedef __attribute__((ext_vector_type(4)))  float    v4f;
typedef __attribute__((ext_vector_type(4)))  int      v4i;

#if __has_builtin(__builtin_amdgcn_global_load_async_to_lds_b128) && \
    __has_builtin(__builtin_amdgcn_s_wait_asynccnt)
#define HAVE_ASYNC_LDS 1
#else
#define HAVE_ASYNC_LDS 0
#endif

#define Bc   16
#define Lc   1024
#define DIN  512
#define DQK  128
#define ROWS (Bc * Lc)          // 16384
#define OUT_ELEMS (ROWS * DIN)  // 8,388,608
#define ATT_PER_HB (Lc * Lc)    // 1,048,576

// workspace offsets (in halves)
#define OFF_LWT   0u            // 3 mats * 2 heads * 128 * 512      = 393216
#define OFF_CWT   393216u       // 3 mats * 2 heads * 3 taps *128*512= 1179648
#define OFF_PROJT 1572864u      // 512 * 512                          = 262144
#define OFF_QH    2097152u      // 4 heads * 16384 * 128              = 8388608
#define OFF_KH    10485760u
#define OFF_VH    18874368u
#define OFF_HOUT  27262976u     // [16384][512] f16
// end: 35651584 halves = 71,303,168 bytes

// ---------------- WMMA fragment helpers (wave32 layouts, ISA 7.12.2) -------

union H16 { v16h v; v8h h[2]; };

// A-frag (16x32, M x K), memory row-major [M][K] (stride in halves).
// lane m=lane&15; chunk0 = k .. k+7 at kbase=(lane>>4)*8; chunk1 = +16.
__device__ inline v16h load_a16(const half_t* base, int stride, int lane) {
  int m = lane & 15;
  int o = (lane >> 4) * 8;
  const half_t* p = base + (size_t)m * stride + o;
  H16 r;
  r.h[0] = *(const v8h*)p;
  r.h[1] = *(const v8h*)(p + 16);
  return r.v;
}

// A-frag from f32 memory with convert-to-f16.
__device__ inline v16h load_a_cvt(const float* base, int stride, int lane) {
  int m = lane & 15;
  int o = (lane >> 4) * 8;
  const float* p = base + (size_t)m * stride + o;
  v16h r;
#pragma unroll
  for (int i = 0; i < 8; ++i) {
    r[i]     = (half_t)p[i];
    r[i + 8] = (half_t)p[i + 16];
  }
  return r;
}

// B-frag (32x16, K x N), memory layout [N][K] (k contiguous, stride in halves).
// lane col n=lane&15; chunk0 = k .. k+7 at kbase=(lane>>4)*8; chunk1 = +16.
__device__ inline v16h load_b16(const half_t* base, int stride, int lane) {
  int n  = lane & 15;
  int kh = (lane >> 4) * 8;
  const half_t* p = base + (size_t)n * stride + kh;
  H16 r;
  r.h[0] = *(const v8h*)p;
  r.h[1] = *(const v8h*)(p + 16);
  return r.v;
}

__device__ inline v8f wmma16(v16h a, v16h b, v8f c) {
  // (neg_a, A, neg_b, B, c_mod, C, reuse_a, reuse_b)
  return __builtin_amdgcn_wmma_f32_16x16x32_f16(false, a, false, b,
                                                (short)0, c, false, false);
}

// ---------------- 1) weight prep: f32 -> f16, transpose to [N][K] ----------

__global__ __launch_bounds__(256)
void prep_weights(const float* lw_q, const float* lw_k, const float* lw_v,
                  const float* cw_q, const float* cw_k, const float* cw_v,
                  const float* proj_w,
                  half_t* lwT, half_t* cwT, half_t* projT) {
  int idx = blockIdx.x * 256 + threadIdx.x;

  if (idx < 393216) {                       // lwT[(mat*2+h)*128 + n][k]
    int k = idx & 511;
    int n = (idx >> 9) & 127;
    int g = idx >> 16;                      // mat*2 + h
    int h = g & 1, mat = g >> 1;
    const float* src = (mat == 0) ? lw_q : (mat == 1) ? lw_k : lw_v;
    lwT[idx] = (half_t)src[((size_t)h * 512 + k) * 128 + n];  // lw [h][K][N]
  }
  int i1 = idx - 393216;
  if (i1 >= 0 && i1 < 1179648) {            // cwT[((mat*2+h)*3+t)*128 + n][k]
    int k = i1 & 511;
    int n = (i1 >> 9) & 127;
    int g = i1 >> 16;                       // (mat*2+h)*3 + t
    int t = g % 3;
    int mh = g / 3;
    int h = mh & 1, mat = mh >> 1;
    const float* src = (mat == 0) ? cw_q : (mat == 1) ? cw_k : cw_v;
    // cw [h][n(out)][k(in)][t]
    cwT[i1] = (half_t)src[(((size_t)h * 128 + n) * 512 + k) * 3 + t];
  }
  int i2 = idx - 1572864;
  if (i2 >= 0 && i2 < 262144) {             // projT[n][k] from proj_w[K][N]
    int k = i2 & 511;
    int n = i2 >> 9;
    projT[i2] = (half_t)proj_w[(size_t)k * 512 + n];
  }
}

// ---------------- 2) learnable projections (6 GEMMs 16384x512 @ 512x128) ---

__global__ __launch_bounds__(256)
void proj_linear(const float* qin, const float* kin, const float* vin,
                 const float* lb_q, const float* lb_k, const float* lb_v,
                 const half_t* lwT, half_t* qh, half_t* kh, half_t* vh) {
  __shared__ half_t lds[32 * 520];          // 32 rows x 512 k, padded stride
  int z = blockIdx.z;                       // mat*2 + h
  int mat = z >> 1, h = z & 1;
  const float* X    = (mat == 0) ? qin : (mat == 1) ? kin : vin;
  const float* bias = ((mat == 0) ? lb_q : (mat == 1) ? lb_k : lb_v) + h * 128;
  half_t* out = ((mat == 0) ? qh : (mat == 1) ? kh : vh)
                + (size_t)h * ROWS * DQK;
  int rows0 = blockIdx.x * 32;
  int tid = threadIdx.x;

  // stage A tile as f16 into LDS (conversion -> must go through VGPRs)
#pragma unroll
  for (int i = 0; i < 64; ++i) {
    int e = tid + i * 256;                  // 32*512 = 16384 elems
    int r = e >> 9, kk = e & 511;
    lds[r * 520 + kk] = (half_t)X[(size_t)(rows0 + r) * 512 + kk];
  }
  __syncthreads();

  int w = tid >> 5, lane = tid & 31;
  int rw = (w & 1) * 16;                    // row tile in LDS
  int cb = (w >> 1) * 32;                   // col base (4 col-waves x 32)
  const half_t* wbase = lwT + (size_t)z * 128 * 512;

  v8f acc[2] = {};
  for (int ks = 0; ks < 16; ++ks) {
    int k0 = ks * 32;
    v16h a = load_a16(lds + rw * 520 + k0, 520, lane);
#pragma unroll
    for (int n = 0; n < 2; ++n) {
      v16h bb = load_b16(wbase + (size_t)(cb + n * 16) * 512 + k0, 512, lane);
      acc[n] = wmma16(a, bb, acc[n]);
    }
  }
  int nn = lane & 15, mo = (lane >> 4) * 8;
#pragma unroll
  for (int n = 0; n < 2; ++n) {
    int col = cb + n * 16 + nn;
    float bv = bias[col];
#pragma unroll
    for (int r = 0; r < 8; ++r)
      out[(size_t)(rows0 + rw + mo + r) * DQK + col] = (half_t)(acc[n][r] + bv);
  }
}

// ---------------- 3) conv1d(k=3,'same') heads as 3 shifted GEMMs -----------

__global__ __launch_bounds__(256)
void proj_conv(const float* qin, const float* kin, const float* vin,
               const float* cb_q, const float* cb_k, const float* cb_v,
               const half_t* cwT, half_t* qh, half_t* kh, half_t* vh) {
  __shared__ half_t lds[34 * 520];          // rows l0-1 .. l0+32
  int z = blockIdx.z;                       // mat*2 + h
  int mat = z >> 1, h = z & 1;
  const float* X    = (mat == 0) ? qin : (mat == 1) ? kin : vin;
  const float* bias = ((mat == 0) ? cb_q : (mat == 1) ? cb_k : cb_v) + h * 128;
  half_t* out = ((mat == 0) ? qh : (mat == 1) ? kh : vh)
                + (size_t)(2 + h) * ROWS * DQK;
  int rows0 = blockIdx.x * 32;              // 1024 % 32 == 0: no batch cross
  int bidx = rows0 >> 10;
  int l0 = rows0 & 1023;
  int tid = threadIdx.x;

#pragma unroll
  for (int i = 0; i < 68; ++i) {            // 34*512 = 17408 = 68*256
    int e = tid + i * 256;
    int r = e >> 9, kk = e & 511;
    int l = l0 + r - 1;
    float xv = (l >= 0 && l < Lc)
                 ? X[((size_t)bidx * Lc + l) * 512 + kk] : 0.0f;
    lds[r * 520 + kk] = (half_t)xv;
  }
  __syncthreads();

  int w = tid >> 5, lane = tid & 31;
  int rw = (w & 1) * 16;
  int cb = (w >> 1) * 32;

  v8f acc[2] = {};
  for (int t = 0; t < 3; ++t) {
    const half_t* wbase = cwT + (size_t)(z * 3 + t) * 128 * 512;
    for (int ks = 0; ks < 16; ++ks) {
      int k0 = ks * 32;
      v16h a = load_a16(lds + (rw + t) * 520 + k0, 520, lane);
#pragma unroll
      for (int n = 0; n < 2; ++n) {
        v16h bb = load_b16(wbase + (size_t)(cb + n * 16) * 512 + k0, 512, lane);
        acc[n] = wmma16(a, bb, acc[n]);
      }
    }
  }
  int nn = lane & 15, mo = (lane >> 4) * 8;
#pragma unroll
  for (int n = 0; n < 2; ++n) {
    int col = cb + n * 16 + nn;
    float bv = bias[col];
#pragma unroll
    for (int r = 0; r < 8; ++r)
      out[(size_t)(rows0 + rw + mo + r) * DQK + col] = (half_t)(acc[n][r] + bv);
  }
}

// ---------------- 4) attention logits: att = Q K^T / sqrt(128) -------------
// K chunk (128 keys x 128 d, f16, 32KB) staged once per workgroup in LDS.
// Async DMA applies a pad (stride 136 halves) so B-frag ds_load_b128s are
// spread across banks instead of the 16-way conflict a 256B stride causes.

__global__ __launch_bounds__(256)
void logits_kernel(const half_t* qh, const half_t* kh, float* att) {
  __shared__ half_t kbuf[128 * 136];        // padded K chunk
  int z = blockIdx.z;                       // h*16 + b
  int h = z >> 4, b = z & 15;
  const half_t* qbase = qh + ((size_t)h * ROWS + (size_t)b * Lc) * DQK;
  const half_t* kbase = kh + ((size_t)h * ROWS + (size_t)b * Lc) * DQK;
  float* abase = att + ((size_t)z << 20);   // z * 1024*1024

  int tid = threadIdx.x, w = tid >> 5, lane = tid & 31;
  int kcb = blockIdx.y * 128;               // staged key base

  // ---- stage K chunk: 128 rows x 16 x 16B = 2048 lane-copies ----
#if HAVE_ASYNC_LDS
#pragma unroll
  for (int i = 0; i < 8; ++i) {
    int e = tid + i * 256;
    int row = e >> 4;
    int c = (e & 15) * 8;                   // half offset within row
    __builtin_amdgcn_global_load_async_to_lds_b128(
        (v4i*)(kbase + (size_t)(kcb + row) * DQK + c),
        (v4i*)(&kbuf[row * 136 + c]),
        0, 0);
  }
  __builtin_amdgcn_s_wait_asynccnt(0);
#else
#pragma unroll
  for (int i = 0; i < 8; ++i) {
    int e = tid + i * 256;
    int row = e >> 4;
    int c = (e & 15) * 8;
    *(v8h*)&kbuf[row * 136 + c] =
        *(const v8h*)(kbase + (size_t)(kcb + row) * DQK + c);
  }
#endif
  __syncthreads();

  int qr  = blockIdx.x * 64 + (w & 3) * 16;
  int kc0 = (w >> 2) * 64;                  // local col base within kbuf

  v16h aq[4];
#pragma unroll
  for (int ks = 0; ks < 4; ++ks)
    aq[ks] = load_a16(qbase + (size_t)qr * DQK + ks * 32, DQK, lane);

  const float inv = 0.08838834764831845f;   // 1/sqrt(128)
  int nn = lane & 15, mo = (lane >> 4) * 8;
#pragma unroll
  for (int n = 0; n < 4; ++n) {
    int kcl = kc0 + n * 16;
    v8f acc = {};
#pragma unroll
    for (int ks = 0; ks < 4; ++ks) {
      v16h bb = load_b16(kbuf + (size_t)kcl * 136 + ks * 32, 136, lane);
      acc = wmma16(aq[ks], bb, acc);
    }
#pragma unroll
    for (int r = 0; r < 8; ++r)
      abase[(size_t)(qr + mo + r) * Lc + kcb + kcl + nn] = acc[r] * inv;
  }
}

// ---------------- 5) row softmax in place (one 1024-row per block) ---------

__global__ __launch_bounds__(256)
void softmax_kernel(float* att) {
  __shared__ float sred[256];
  size_t row = blockIdx.x;
  float* p = att + row * Lc;
  int tid = threadIdx.x;

  v4f x = *((const v4f*)p + tid);
  float m = fmaxf(fmaxf(x[0], x[1]), fmaxf(x[2], x[3]));
  sred[tid] = m;
  __syncthreads();
  for (int s = 128; s > 0; s >>= 1) {
    if (tid < s) sred[tid] = fmaxf(sred[tid], sred[tid + s]);
    __syncthreads();
  }
  float mx = sred[0];
  __syncthreads();

  v4f e;
  float sum = 0.0f;
#pragma unroll
  for (int i = 0; i < 4; ++i) { e[i] = __expf(x[i] - mx); sum += e[i]; }
  sred[tid] = sum;
  __syncthreads();
  for (int s = 128; s > 0; s >>= 1) {
    if (tid < s) sred[tid] += sred[tid + s];
    __syncthreads();
  }
  float rinv = 1.0f / sred[0];
  e = e * rinv;
  *((v4f*)p + tid) = e;
}

// ---------------- 6) out_head = scale * att @ V ----------------------------

__global__ __launch_bounds__(256)
void attv_kernel(const float* att, const half_t* vh,
                 const float* lscale, const float* cscale, half_t* hout) {
  __shared__ half_t vT[128 * 40];           // V chunk transposed [d][m(32)]
  int z = blockIdx.z;                       // h*16 + b
  int h = z >> 4, b = z & 15;
  const float* abase = att + ((size_t)z << 20);
  const half_t* vbase = vh + ((size_t)h * ROWS + (size_t)b * Lc) * DQK;
  int tid = threadIdx.x, w = tid >> 5, lane = tid & 31;
  int lrow = blockIdx.x * 128 + w * 16;     // this wave's 16 output rows

  v8f acc[8] = {};
  for (int mc = 0; mc < 32; ++mc) {         // k(=m) chunks of 32
    {                                       // cooperative transpose-stage of V
      int e = tid * 16;                     // 32*128 halves / 256 thr = 16 ea
      int mi = e >> 7;
      int d0 = e & 127;
      const half_t* src = vbase + (size_t)(mc * 32 + mi) * DQK + d0;
#pragma unroll
      for (int j = 0; j < 16; ++j) vT[(d0 + j) * 40 + mi] = src[j];
    }
    __syncthreads();
    if (mc + 1 < 32)                        // pull next att chunk ahead
      __builtin_prefetch(abase + (size_t)lrow * Lc + (mc + 1) * 32, 0, 0);
    v16h a = load_a_cvt(abase + (size_t)lrow * Lc + mc * 32, Lc, lane);
#pragma unroll
    for (int n = 0; n < 8; ++n) {
      v16h bb = load_b16(vT + (size_t)(n * 16) * 40, 40, lane);
      acc[n] = wmma16(a, bb, acc[n]);
    }
    __syncthreads();
  }

  float scale = (h < 2) ? lscale[h] : cscale[h - 2];
  int nn = lane & 15, mo = (lane >> 4) * 8;
#pragma unroll
  for (int n = 0; n < 8; ++n) {
    int col = h * 128 + n * 16 + nn;
#pragma unroll
    for (int r = 0; r < 8; ++r)
      hout[(size_t)(b * Lc + lrow + mo + r) * 512 + col] =
          (half_t)(acc[n][r] * scale);
  }
}

// ---------------- 7) final projection: out = hout @ proj_w + proj_b --------

__global__ __launch_bounds__(256)
void final_proj(const half_t* hout, const half_t* projT,
                const float* proj_b, float* out) {
  int tid = threadIdx.x, w = tid >> 5, lane = tid & 31;
  int rows0 = blockIdx.x * 64 + (w & 3) * 16;
  int cb    = blockIdx.y * 128 + (w >> 2) * 64;

  v8f acc[4] = {};
  for (int ks = 0; ks < 16; ++ks) {
    int k0 = ks * 32;
    if (ks + 1 < 16)
      __builtin_prefetch(hout + (size_t)rows0 * 512 + k0 + 32, 0, 0);
    v16h a = load_a16(hout + (size_t)rows0 * 512 + k0, 512, lane);
#pragma unroll
    for (int n = 0; n < 4; ++n) {
      v16h bb = load_b16(projT + (size_t)(cb + n * 16) * 512 + k0, 512, lane);
      acc[n] = wmma16(a, bb, acc[n]);
    }
  }
  int nn = lane & 15, mo = (lane >> 4) * 8;
#pragma unroll
  for (int n = 0; n < 4; ++n) {
    int col = cb + n * 16 + nn;
    float bv = proj_b[col];
#pragma unroll
    for (int r = 0; r < 8; ++r)
      out[(size_t)(rows0 + mo + r) * 512 + col] = acc[n][r] + bv;
  }
}

// ---------------- host-side launcher ---------------------------------------

extern "C" void kernel_launch(void* const* d_in, const int* in_sizes, int n_in,
                              void* d_out, int out_size, void* d_ws,
                              size_t ws_size, hipStream_t stream) {
  (void)in_sizes; (void)n_in; (void)out_size; (void)ws_size;

  const float* query  = (const float*)d_in[0];
  const float* key_   = (const float*)d_in[1];
  const float* value  = (const float*)d_in[2];
  const float* lw_q   = (const float*)d_in[3];
  const float* lb_q   = (const float*)d_in[4];
  const float* lw_k   = (const float*)d_in[5];
  const float* lb_k   = (const float*)d_in[6];
  const float* lw_v   = (const float*)d_in[7];
  const float* lb_v   = (const float*)d_in[8];
  const float* lscale = (const float*)d_in[9];
  const float* cw_q   = (const float*)d_in[10];
  const float* cb_q   = (const float*)d_in[11];
  const float* cw_k   = (const float*)d_in[12];
  const float* cb_k   = (const float*)d_in[13];
  const float* cw_v   = (const float*)d_in[14];
  const float* cb_v   = (const float*)d_in[15];
  const float* cscale = (const float*)d_in[16];
  const float* proj_w = (const float*)d_in[17];
  const float* proj_b = (const float*)d_in[18];

  float* out = (float*)d_out;
  float* att = out + OUT_ELEMS;             // 4 x [16,1024,1024] f32

  half_t* wsh   = (half_t*)d_ws;            // needs ~71.3 MB
  half_t* lwT   = wsh + OFF_LWT;
  half_t* cwT   = wsh + OFF_CWT;
  half_t* projT = wsh + OFF_PROJT;
  half_t* qh    = wsh + OFF_QH;
  half_t* kh    = wsh + OFF_KH;
  half_t* vh    = wsh + OFF_VH;
  half_t* hout  = wsh + OFF_HOUT;

  prep_weights<<<7168, 256, 0, stream>>>(lw_q, lw_k, lw_v, cw_q, cw_k, cw_v,
                                         proj_w, lwT, cwT, projT);
  proj_linear<<<dim3(512, 1, 6), 256, 0, stream>>>(
      query, key_, value, lb_q, lb_k, lb_v, lwT, qh, kh, vh);
  proj_conv<<<dim3(512, 1, 6), 256, 0, stream>>>(
      query, key_, value, cb_q, cb_k, cb_v, cwT, qh, kh, vh);
  logits_kernel<<<dim3(16, 8, 64), 256, 0, stream>>>(qh, kh, att);
  softmax_kernel<<<65536, 256, 0, stream>>>(att);
  attv_kernel<<<dim3(8, 1, 64), 256, 0, stream>>>(att, vh, lscale, cscale,
                                                  hout);
  final_proj<<<dim3(256, 4, 1), 256, 0, stream>>>(hout, projT, proj_b, out);
}
